// MPN_19662360281744
// MI455X (gfx1250) — matrix operand order repaired and checked by reading, not compiled, not run on hardware
//
#include <hip/hip_runtime.h>

// Message-passing network, fused per-edge pipeline on CDNA5 WMMA (gfx1250).
// d_out = [ nm (100000*32 f32) | edge_msg (3200000*6 f32) ]
// be1/bn1 folded into weight rows (K=70 / K=38, driven by constant-1.0 A inputs);
// be2/bn2 added as scalar v_add before relu. All accumulators start from the
// inline-0 C operand. Registers kept < 256 (no s_set_vgpr_msb mode).

#define E_TOTAL     3200000
#define NN          100000
#define TILES       (E_TOTAL / 16)          // 200000, exact
#define WAVES_PB    8
#define BLOCK_T     256
#define NUM_BLOCKS  2048

typedef __attribute__((ext_vector_type(16))) __bf16 v16bf;
typedef __attribute__((ext_vector_type(8)))  float  v8f;

#define DI __device__ __forceinline__

DI v8f wmma_bf16(v16bf a, v16bf b, v8f c) {
  // D = A(16x32 bf16) x B(32x16 bf16) + C(16x16 f32)
  return __builtin_amdgcn_wmma_f32_16x16x32_bf16(
      /*neg_a=*/false, a, /*neg_b=*/false, b,
      /*c_mod=*/(short)0, c, /*reuse_a=*/false, /*reuse_b=*/false);
}

DI void lds_fence_wave() {
  // LDS ops are in-order per wave; this forces completion + blocks compiler reordering.
  asm volatile("s_wait_dscnt 0x0" ::: "memory");
}

DI float relu1(float x) {
  // Single VOP2 max with inline-constant 0; un-foldable (no canonicalize pair).
  float y;
  asm("v_max_num_f32 %0, 0, %1" : "=v"(y) : "v"(x));
  return y;
}

DI void relu8(v8f& v) {
#pragma unroll
  for (int i = 0; i < 8; ++i) v[i] = relu1(v[i]);
}

DI void addrelu8(v8f& v, float b) {
#pragma unroll
  for (int i = 0; i < 8; ++i) v[i] = relu1(v[i] + b);
}

// B fragment (32x16 tile of a row-major [K x nstride] bf16 LDS array).
// B 32x16 layout matching A: col N=lane&15, khalf = (lane<16)?0:8;
// element i<8 -> K = khalf + i, element i>=8 -> K = 16 + khalf + (i-8).
DI v16bf b_frag(const __bf16* W, int nstride, int kbase, int nbase, int lane) {
  int col   = nbase + (lane & 15);
  int khalf = (lane < 16) ? 0 : 8;
  v16bf b;
#pragma unroll
  for (int i = 0; i < 16; ++i) b[i] = W[(kbase + khalf + (i & 7) + ((i >> 3) << 4)) * nstride + col];
  return b;
}

// A fragment (16 x 32 bf16) from a row-major f32 LDS staging tile.
// ISA 16-bit A 16x32 layout: lane L row M=L&15; elements 0..7 -> K=koff..koff+7,
// elements 8..15 -> K=16+koff..16+koff+7, koff = (L<16)?0:8.
DI v16bf a_frag_lds(const float* S, int kstride, int kbase, int lane) {
  int row  = lane & 15;
  int koff = (lane < 16) ? 0 : 8;
  const float* p = S + row * kstride + kbase + koff;
  v16bf a;
#pragma unroll
  for (int i = 0; i < 8; ++i) {
    a[i]     = (__bf16)p[i];
    a[i + 8] = (__bf16)p[16 + i];
  }
  return a;
}

// A fragment for a gathered node row (32 contiguous f32 in global, 128b loads).
DI v16bf a_frag_node(const float* __restrict__ xn, int idx, int lane) {
  int koff = (lane < 16) ? 0 : 8;
  const float* p = xn + (size_t)idx * 32 + koff;
  float4 u0 = *(const float4*)(p);
  float4 u1 = *(const float4*)(p + 4);
  float4 w0 = *(const float4*)(p + 16);
  float4 w1 = *(const float4*)(p + 20);
  v16bf a;
  a[0]  = (__bf16)u0.x; a[1]  = (__bf16)u0.y; a[2]  = (__bf16)u0.z; a[3]  = (__bf16)u0.w;
  a[4]  = (__bf16)u1.x; a[5]  = (__bf16)u1.y; a[6]  = (__bf16)u1.z; a[7]  = (__bf16)u1.w;
  a[8]  = (__bf16)w0.x; a[9]  = (__bf16)w0.y; a[10] = (__bf16)w0.z; a[11] = (__bf16)w0.w;
  a[12] = (__bf16)w1.x; a[13] = (__bf16)w1.y; a[14] = (__bf16)w1.z; a[15] = (__bf16)w1.w;
  return a;
}

// A fragment for x_edge: 6 real K values, K=6 carries constant 1.0 (be1 folded
// into weight row 70), K=7..31 zero. Only lanes 0-15 hold K<8.
DI v16bf a_frag_edge(const float* __restrict__ xe, int e, int lane) {
  v16bf a;
#pragma unroll
  for (int i = 0; i < 16; ++i) a[i] = (__bf16)0.0f;
  if (lane < 16) {
    const float* p = xe + (size_t)e * 6;
#pragma unroll
    for (int i = 0; i < 6; ++i) a[i] = (__bf16)p[i];
    a[6] = (__bf16)1.0f;   // bias input (overall K = 70)
  }
  return a;
}

__global__ void __launch_bounds__(BLOCK_T)
mpn_zero_nm(float4* __restrict__ p, int n4) {
  int i = blockIdx.x * BLOCK_T + threadIdx.x;
  if (i < n4) {
    float4 z; z.x = 0.f; z.y = 0.f; z.z = 0.f; z.w = 0.f;
    p[i] = z;
  }
}

__global__ void __launch_bounds__(BLOCK_T)
mpn_fused(const float* __restrict__ x_node, const float* __restrict__ x_edge,
          const int* __restrict__ src, const int* __restrict__ dst,
          const float* __restrict__ We1, const float* __restrict__ be1,
          const float* __restrict__ We2, const float* __restrict__ be2,
          const float* __restrict__ Wn1, const float* __restrict__ bn1,
          const float* __restrict__ Wn2, const float* __restrict__ bn2,
          float* __restrict__ nm_out, float* __restrict__ em_out) {
  // ---- LDS: bf16 weights (K padded to mult of 32, N padded to mult of 16) ----
  __shared__ __bf16 sWe1[96 * 32];   // edge in 70->96 pad; row 70 = be1
  __shared__ __bf16 sWe2[32 * 16];   // 32 -> 6 (N pad 16)
  __shared__ __bf16 sWn1[64 * 64];   // node in 38->64 pad; row 38 = bn1
  __shared__ __bf16 sWn2[64 * 32];   // 64 -> 32
  __shared__ float  sBe2[16], sBn2[32];
  __shared__ __align__(16) float sSH[WAVES_PB][16 * 64];  // per-wave h staging
  __shared__ __align__(16) float sSM[WAVES_PB][16 * 32];  // per-wave edge_msg staging
                                                          // (col 6 == 1.0 bias input)

  const int tid  = threadIdx.x;
  const int lane = tid & 31;
  const int wv   = tid >> 5;

  // ---- cooperative weight preload (f32 -> bf16, zero-padded, bias rows) ----
  for (int i = tid; i < 96 * 32; i += BLOCK_T) {
    int k = i >> 5, n = i & 31;
    float v = (k < 70) ? We1[k * 32 + n] : (k == 70 ? be1[n] : 0.0f);
    sWe1[i] = (__bf16)v;
  }
  for (int i = tid; i < 32 * 16; i += BLOCK_T) {
    int k = i >> 4, n = i & 15;
    sWe2[i] = (__bf16)((n < 6) ? We2[k * 6 + n] : 0.0f);
  }
  for (int i = tid; i < 64 * 64; i += BLOCK_T) {
    int k = i >> 6, n = i & 63;
    float v = (k < 38) ? Wn1[k * 64 + n] : (k == 38 ? bn1[n] : 0.0f);
    sWn1[i] = (__bf16)v;
  }
  for (int i = tid; i < 64 * 32; i += BLOCK_T) sWn2[i] = (__bf16)Wn2[i];
  if (tid < 16) sBe2[tid] = (tid < 6) ? be2[tid] : 0.0f;
  if (tid < 32) sBn2[tid] = bn2[tid];
  // edge_msg staging: zeros except col 6 == 1.0 (bias input for Wn1 row 38);
  // per-tile stores only touch cols 0..5, so this persists across the loop.
  for (int i = tid; i < WAVES_PB * 16 * 32; i += BLOCK_T) {
    (&sSM[0][0])[i] = ((i & 31) == 6) ? 1.0f : 0.0f;
  }
  __syncthreads();

  // ---- hoist all 19 B fragments into registers (152 VGPRs, loop-invariant) ----
  v16bf Be1[3][2];
#pragma unroll
  for (int kc = 0; kc < 3; ++kc)
#pragma unroll
    for (int nt = 0; nt < 2; ++nt) Be1[kc][nt] = b_frag(sWe1, 32, kc * 32, nt * 16, lane);
  v16bf Be2f = b_frag(sWe2, 16, 0, 0, lane);
  v16bf Bn1[2][4];
#pragma unroll
  for (int kc = 0; kc < 2; ++kc)
#pragma unroll
    for (int nt = 0; nt < 4; ++nt) Bn1[kc][nt] = b_frag(sWn1, 64, kc * 32, nt * 16, lane);
  v16bf Bn2[2][2];
#pragma unroll
  for (int kc = 0; kc < 2; ++kc)
#pragma unroll
    for (int nt = 0; nt < 2; ++nt) Bn2[kc][nt] = b_frag(sWn2, 32, kc * 32, nt * 16, lane);

  const int col  = lane & 15;
  const int moff = (lane < 16) ? 0 : 8;

  // per-lane bias values for the layer-2 GEMMs (3 VGPRs, not 4 fragments)
  const float bvE2  = sBe2[col];
  const float bvN20 = sBn2[col];
  const float bvN21 = sBn2[16 + col];

  float* SH = sSH[wv];
  float* SM = sSM[wv];

  const int waveG    = blockIdx.x * WAVES_PB + wv;
  const int numWaves = NUM_BLOCKS * WAVES_PB;

  for (int t = waveG; t < TILES; t += numWaves) {
    const int e0  = t * 16;
    const int row = lane & 15;
    const int myDst = dst[e0 + row];
    const int mySrc = src[e0 + row];

    // gathered A fragments (x_node table is L2-resident: 12.8 MB << 192 MB)
    v16bf aD = a_frag_node(x_node, myDst, lane);
    v16bf aS = a_frag_node(x_node, mySrc, lane);
    v16bf aE = a_frag_edge(x_edge, e0 + row, lane);

    const v8f zf = {};   // inline-0 C operand

    // ---- edge layer 1: h_e[16x32] = relu([xd|xs|xe|1] @ [We1;be1]) ----
    lds_fence_wave();    // SH WAR vs previous iteration
#pragma unroll
    for (int nt = 0; nt < 2; ++nt) {
      v8f acc = wmma_bf16(aD, Be1[0][nt], zf);
      acc = wmma_bf16(aS, Be1[1][nt], acc);
      acc = wmma_bf16(aE, Be1[2][nt], acc);
      relu8(acc);
#pragma unroll
      for (int r = 0; r < 8; ++r) SH[(r + moff) * 32 + nt * 16 + col] = acc[r];
    }
    lds_fence_wave();
    v16bf aH = a_frag_lds(SH, 32, 0, lane);

    // ---- edge layer 2: edge_msg[16x6] = relu(h_e @ We2 + be2) ----
    v8f accM = wmma_bf16(aH, Be2f, zf);
    addrelu8(accM, bvE2);                   // padded cols 6..15 stay exact 0

    lds_fence_wave();    // SM readers of previous iteration are done
    if (col < 6) {       // keep col 6 == 1.0 and cols 7..31 == 0
#pragma unroll
      for (int r = 0; r < 8; ++r) SM[(r + moff) * 32 + col] = accM[r];
    }
    lds_fence_wave();
    v16bf aM = a_frag_lds(SM, 32, 0, lane); // edge_msg | 1.0 as A chunk

    // coalesced edge_msg output: 16 edges x 6 f32 = 96 contiguous floats
    {
      float* oe = em_out + (size_t)t * 96;
#pragma unroll
      for (int j = 0; j < 3; ++j) {
        int f = lane * 3 + j;
        oe[f] = SM[(f / 6) * 32 + (f % 6)];
      }
    }
    lds_fence_wave();    // done reading SM/SH before h_n staging overwrites SH

    // ---- node layer 1: h_n[16x64] = relu([xd|edge_msg|1] @ [Wn1;bn1]) ----
#pragma unroll
    for (int nt = 0; nt < 4; ++nt) {
      v8f acc = wmma_bf16(aD, Bn1[0][nt], zf);
      acc = wmma_bf16(aM, Bn1[1][nt], acc);
      relu8(acc);
#pragma unroll
      for (int r = 0; r < 8; ++r) SH[(r + moff) * 64 + nt * 16 + col] = acc[r];
    }
    lds_fence_wave();
    v16bf aH0 = a_frag_lds(SH, 64, 0, lane);
    v16bf aH1 = a_frag_lds(SH, 64, 32, lane);

    // ---- node layer 2 + scatter: nm[dst] += relu(h_n @ Wn2 + bn2) ----
#pragma unroll
    for (int nt = 0; nt < 2; ++nt) {
      v8f acc = wmma_bf16(aH0, Bn2[0][nt], zf);
      acc = wmma_bf16(aH1, Bn2[1][nt], acc);
      addrelu8(acc, nt == 0 ? bvN20 : bvN21);
#pragma unroll
      for (int r = 0; r < 8; ++r) {
        // broadcast load of dst[e0 + M] (2 distinct L0-hit addresses per r);
        // keeps this off the busy LDS pipe (vs ds_bpermute shuffles)
        int d = dst[e0 + r + moff];
        float* p = nm_out + (size_t)d * 32 + nt * 16 + col;
        __hip_atomic_fetch_add(p, acc[r], __ATOMIC_RELAXED, __HIP_MEMORY_SCOPE_AGENT);
      }
    }
  }
}

extern "C" void kernel_launch(void* const* d_in, const int* in_sizes, int n_in,
                              void* d_out, int out_size, void* d_ws, size_t ws_size,
                              hipStream_t stream) {
  (void)in_sizes; (void)n_in; (void)out_size; (void)d_ws; (void)ws_size;
  const float* x_node = (const float*)d_in[0];
  const float* x_edge = (const float*)d_in[1];
  const int*   src    = (const int*)d_in[2];
  const int*   dst    = (const int*)d_in[3];
  const float* We1 = (const float*)d_in[4];  const float* be1 = (const float*)d_in[5];
  const float* We2 = (const float*)d_in[6];  const float* be2 = (const float*)d_in[7];
  const float* Wn1 = (const float*)d_in[8];  const float* bn1 = (const float*)d_in[9];
  const float* Wn2 = (const float*)d_in[10]; const float* bn2 = (const float*)d_in[11];

  float* nm = (float*)d_out;                       // [100000 x 32]
  float* em = nm + (size_t)NN * 32;                // [3200000 x 6]

  // d_out is poisoned by the harness: zero the atomic-accumulated region.
  const int n4 = (NN * 32) / 4;                    // 800000 float4
  mpn_zero_nm<<<(n4 + BLOCK_T - 1) / BLOCK_T, BLOCK_T, 0, stream>>>((float4*)nm, n4);

  mpn_fused<<<NUM_BLOCKS, BLOCK_T, 0, stream>>>(
      x_node, x_edge, src, dst, We1, be1, We2, be2, Wn1, bn1, Wn2, bn2, nm, em);
}
